// BiLSTM_CRF_40046275068214
// MI455X (gfx1250) — matrix operand order repaired and checked by reading, not compiled, not run on hardware
//
#include <hip/hip_runtime.h>
#include <hip/hip_bf16.h>
#include <math.h>

// BiLSTM-CRF for MI455X (gfx1250, wave32, WMMA).
//  - x@W_ih^T hoisted out of recurrences: M=32768,N=512,K=256 GEMMs on
//    v_wmma_f32_16x16x32_f16 (HBM-bound, ~13us floor at 23.3 TB/s).
//  - Recurrence h@W_hh^T: 1 workgroup/direction, W_hh in LDS (128KB/320KB),
//    h double-buffered in LDS, c in registers; each wave owns the same 16
//    hidden units in all 4 gate blocks -> register-local LSTM epilogue with
//    fast exp-based sigmoid/tanh (v_exp_f32 + fast rcp).
//  - Viterbi: 12 states spread over 16-lane groups (shfl max-plus, 12x shorter
//    serial chain), backpointers in LDS (96KB/block) so the 512-step backtrace
//    chases pointers at LDS latency, not L2 latency.

typedef __attribute__((ext_vector_type(16))) _Float16 v16h;
typedef __attribute__((ext_vector_type(8)))  _Float16 v8h;
typedef __attribute__((ext_vector_type(8)))  float    v8f;

#define LL   512
#define BB   64
#define EE   256
#define HH   256
#define HD   128      // H2
#define G4   512      // 4*H2 gate width
#define LB   (LL*BB)  // 32768 rows
#define CCT  12
#define START_TAG 10
#define STOP_TAG  11
#define IMPOSSIBLE_F (-10000.0f)

static __device__ __forceinline__ v16h load_a16(const _Float16* p) {
  // A-matrix fragment: 16 contiguous halves (two adjacent 16B chunks) per lane.
  union { v16h v; v8h h[2]; } u;
  u.h[0] = *(const v8h*)(p);
  u.h[1] = *(const v8h*)(p + 8);
  return u.v;
}

static __device__ __forceinline__ float sigf(float x) {
  return __fdividef(1.0f, 1.0f + __expf(-x));
}
static __device__ __forceinline__ float tanh_fast(float x) {
  // tanh(x) = 2*sigmoid(2x) - 1 : one v_exp_f32 + one fast rcp
  return __fdividef(2.0f, 1.0f + __expf(-2.0f * x)) - 1.0f;
}

// ---------------------------------------------------------------- weights->f16
__global__ void convert_weights(const float* wihf1, const float* wihr1,
                                const float* wihf2, const float* wihr2,
                                const float* whhf1, const float* whhr1,
                                const float* whhf2, const float* whhr2,
                                _Float16* WihH, _Float16* WhhH) {
  const int NIH = G4 * EE;   // 131072
  const int NHH = G4 * HD;   // 65536
  int idx = blockIdx.x * blockDim.x + threadIdx.x;
  if (idx < 4 * NIH) {
    int which = idx / NIH, off = idx % NIH;
    const float* s = which == 0 ? wihf1 : which == 1 ? wihr1 : which == 2 ? wihf2 : wihr2;
    WihH[(size_t)which * NIH + off] = (_Float16)s[off];
  } else {
    int r = idx - 4 * NIH;
    if (r < 4 * NHH) {
      int which = r / NHH, off = r % NHH;
      const float* s = which == 0 ? whhf1 : which == 1 ? whhr1 : which == 2 ? whhf2 : whhr2;
      WhhH[(size_t)which * NHH + off] = (_Float16)s[off];
    }
  }
}

// ------------------------------------------------------------------- embedding
// embeds flat row r (of the (L,B) reshape) = emb_table[sentence.flat[r]]
__global__ void embed_kernel(const int* __restrict__ sent,
                             const float* __restrict__ table,
                             _Float16* __restrict__ Xh) {
  int row = blockIdx.x;     // 0..LB-1
  int col = threadIdx.x;    // 0..255
  int tok = sent[row];
  Xh[(size_t)row * EE + col] = (_Float16)table[(size_t)tok * EE + col];
}

// --------------------------------------------------- gates GEMM: G = X*W^T + b
// M=LB, N=512, K=256. One block = 8 waves, each wave -> 16(M) x 64(N) strip.
__global__ void gates_gemm(const _Float16* __restrict__ X,
                           const _Float16* __restrict__ Wf,
                           const _Float16* __restrict__ Wr,
                           const float* __restrict__ bihf, const float* __restrict__ bhhf,
                           const float* __restrict__ bihr, const float* __restrict__ bhhr,
                           float* __restrict__ Gf, float* __restrict__ Gr) {
  const int dir = blockIdx.z;
  const _Float16* W = dir ? Wr : Wf;
  const float* bih  = dir ? bihr : bihf;
  const float* bhh  = dir ? bhhr : bhhf;
  float* G          = dir ? Gr : Gf;

  const int lane = threadIdx.x & 31;
  const int wv   = threadIdx.x >> 5;
  const int lhi  = lane >> 4, llo = lane & 15;
  const int rowBase = (blockIdx.x * 8 + wv) * 16;
  const int nBase   = blockIdx.y * 64;

  const _Float16* rowp = X + (size_t)(rowBase + llo) * EE;
  v8f acc[4] = {};
#pragma unroll
  for (int kt = 0; kt < 8; ++kt) {          // K = 256 = 8 * 32
    // prefetch the streamed A operand two k-steps ahead (global_prefetch_b8)
    __builtin_prefetch(rowp + kt * 32 + 64, 0, 0);
    v16h a = load_a16(rowp + kt * 32 + lhi * 16);
#pragma unroll
    for (int i = 0; i < 4; ++i) {
      int n = nBase + i * 16 + llo;         // B-lane holds W row n (contig 32B)
      v16h b = *(const v16h*)(W + (size_t)n * EE + kt * 32 + lhi * 16);
      acc[i] = __builtin_amdgcn_wmma_f32_16x16x32_f16(
          false, a, false, b, (short)0, acc[i], false, false);
    }
  }
#pragma unroll
  for (int i = 0; i < 4; ++i) {
    int n = nBase + i * 16 + llo;
    float bias = bih[n] + bhh[n];
#pragma unroll
    for (int r = 0; r < 8; ++r) {
      int row = rowBase + r + lhi * 8;
      G[(size_t)row * G4 + n] = acc[i][r] + bias;
    }
  }
}

// -------------------------------------------------------- recurrent LSTM chain
// grid = 2 blocks (dir 0 fwd, 1 rev), 256 threads = 8 waves.
// Wave w owns hidden units j in [16w,16w+16) across ALL four gate blocks.
__global__ void lstm_rec(const float* __restrict__ Gf, const float* __restrict__ Gr,
                         const _Float16* __restrict__ WhhFh, const _Float16* __restrict__ WhhRh,
                         const float* __restrict__ h0, const float* __restrict__ c0,
                         _Float16* __restrict__ Out) {
  extern __shared__ _Float16 smem[];
  _Float16* sW  = smem;                       // 512*128 halves = 128KB
  _Float16* sHa = smem + G4 * HD;             // 64*128 halves
  _Float16* sHb = sHa + BB * HD;              // 64*128 halves

  const int dir = blockIdx.x;
  const float* G       = dir ? Gr : Gf;
  const _Float16* Wg   = dir ? WhhRh : WhhFh;
  const float* h0p     = h0 + (size_t)dir * BB * HD;
  const float* c0p     = c0 + (size_t)dir * BB * HD;

  const int tid = threadIdx.x, lane = tid & 31, wv = tid >> 5;
  const int lhi = lane >> 4, llo = lane & 15;
  const int j = wv * 16 + llo;                // this lane's hidden unit

  for (int i = tid; i < G4 * HD / 2; i += 256)
    ((unsigned int*)sW)[i] = ((const unsigned int*)Wg)[i];
  for (int i = tid; i < BB * HD; i += 256)
    sHa[i] = (_Float16)h0p[i];

  float c[4][8];
#pragma unroll
  for (int mt = 0; mt < 4; ++mt)
#pragma unroll
    for (int r = 0; r < 8; ++r) {
      int b = mt * 16 + r + lhi * 8;
      c[mt][r] = c0p[b * HD + j];
    }
  __syncthreads();

  for (int t = 0; t < LL; ++t) {
    const int tt = dir ? (LL - 1 - t) : t;
    const _Float16* hcur = (t & 1) ? sHb : sHa;
    _Float16*       hnxt = (t & 1) ? sHa : sHb;

    // prefetch next timestep's precomputed gates while this step computes
    if (t + 1 < LL) {
      int tn = dir ? (LL - 2 - t) : (t + 1);
      __builtin_prefetch(&G[((size_t)tn * BB + lhi * 8) * G4 + j], 0, 0);
    }

#pragma unroll 1
    for (int mt = 0; mt < 4; ++mt) {
      v8f acc[4] = {};
#pragma unroll
      for (int kt = 0; kt < 4; ++kt) {        // K = 128 = 4 * 32
        v16h a = load_a16(hcur + (mt * 16 + llo) * HD + kt * 32 + lhi * 16);
#pragma unroll
        for (int gi = 0; gi < 4; ++gi) {
          int n = gi * HD + j;                // gate gi, hidden unit j
          v16h b = *(const v16h*)(sW + n * HD + kt * 32 + lhi * 16);
          acc[gi] = __builtin_amdgcn_wmma_f32_16x16x32_f16(
              false, a, false, b, (short)0, acc[gi], false, false);
        }
      }
#pragma unroll
      for (int r = 0; r < 8; ++r) {
        int b = mt * 16 + r + lhi * 8;
        size_t gbase = ((size_t)tt * BB + b) * G4 + j;
        float g_i = acc[0][r] + G[gbase + 0 * HD];
        float g_f = acc[1][r] + G[gbase + 1 * HD];
        float g_g = acc[2][r] + G[gbase + 2 * HD];
        float g_o = acc[3][r] + G[gbase + 3 * HD];
        float cc = sigf(g_f) * c[mt][r] + sigf(g_i) * tanh_fast(g_g);
        c[mt][r] = cc;
        float h = sigf(g_o) * tanh_fast(cc);
        hnxt[b * HD + j] = (_Float16)h;
        Out[((size_t)tt * BB + b) * HH + dir * HD + j] = (_Float16)h;
      }
    }
    __syncthreads();
  }
}

// --------------------------------------------------------- projection to tags
__global__ void proj_kernel(const _Float16* __restrict__ X2,
                            const float* __restrict__ wout,
                            const float* __restrict__ bout,
                            float* __restrict__ F) {
  const int lane = threadIdx.x & 31, wv = threadIdx.x >> 5;
  const int row = blockIdx.x * 8 + wv;
  float x[8];
  v8h xv = *(const v8h*)(X2 + (size_t)row * HH + lane * 8);
#pragma unroll
  for (int k = 0; k < 8; ++k) x[k] = (float)xv[k];
#pragma unroll
  for (int ci = 0; ci < CCT; ++ci) {
    float p = 0.f;
#pragma unroll
    for (int k = 0; k < 8; ++k) p += x[k] * wout[ci * HH + lane * 8 + k];
#pragma unroll
    for (int off = 16; off >= 1; off >>= 1) p += __shfl_down(p, off, 32);
    if (lane == 0) F[(size_t)row * CCT + ci] = p + bout[ci];
  }
}

// ----------------------------------------------------- Viterbi fwd + backtrace
// 4 blocks x 256 threads. Each 16-lane group handles one batch element; lane c
// owns CRF state c (c<12 active). Backpointers live in LDS (512*16*12 = 96KB).
__global__ void viterbi_kernel(const float* __restrict__ F,
                               const float* __restrict__ masks,
                               const float* __restrict__ trans,
                               float* __restrict__ out) {
  extern __shared__ unsigned char sBP[];     // [LL][16][CCT]
  const int tid = threadIdx.x;
  const int c = tid & 15;                    // state owned by this lane
  const int b_local = tid >> 4;              // 0..15
  const int b = blockIdx.x * 16 + b_local;
  const bool act = (c < CCT);

  float trrow[CCT];
#pragma unroll
  for (int p = 0; p < CCT; ++p) trrow[p] = act ? trans[c * CCT + p] : 0.0f;

  float ms = (act && c == START_TAG) ? 0.0f : IMPOSSIBLE_F;

  // sequence length: strided partial sums, reduce within the 16-lane group
  float lpart = 0.f;
  for (int t = c; t < LL; t += 16) lpart += masks[(size_t)b * LL + t];
#pragma unroll
  for (int off = 8; off >= 1; off >>= 1) lpart += __shfl_down(lpart, off, 16);
  const int len = (int)__shfl(lpart, 0, 16);

  for (int t = 0; t < LL; ++t) {
    float m = masks[(size_t)b * LL + t];
    float emit = act ? F[((size_t)b * LL + t) * CCT + c] : 0.0f;
    float best = -3.4e38f; int arg = 0;
#pragma unroll
    for (int p = 0; p < CCT; ++p) {          // broadcast ms[p] across the group
      float v = __shfl(ms, p, 16) + trrow[p];
      if (v > best) { best = v; arg = p; }
    }
    if (act) sBP[((size_t)t * 16 + b_local) * CCT + c] = (unsigned char)arg;
    float sc = best + emit;
    ms = sc * m + ms * (1.0f - m);
  }

  float fin = act ? (ms + trans[STOP_TAG * CCT + c]) : -3.4e38f;
  // group-wide argmax (all lanes active so shfl sources are valid)
  float best = -3.4e38f; int btag = 0;
#pragma unroll
  for (int p = 0; p < CCT; ++p) {
    float v = __shfl(fin, p, 16);
    if (v > best) { best = v; btag = p; }
  }
  __syncthreads();

  if (c == 0) {
    out[b] = best;
    int cur = btag;
    for (int t = LL - 1; t >= 0; --t) {
      int tn = (t == LL - 1) ? t : (t + 1);  // bps_next
      int nxt = sBP[((size_t)tn * 16 + b_local) * CCT + cur];
      int nv = (t >= len - 1) ? btag : nxt;
      cur = nv;
      out[BB + (size_t)b * LL + t] = (t < len) ? (float)nv : 0.0f;
    }
  }
}

// ---------------------------------------------------------------------- launch
extern "C" void kernel_launch(void* const* d_in, const int* in_sizes, int n_in,
                              void* d_out, int out_size, void* d_ws, size_t ws_size,
                              hipStream_t stream) {
  const int*   sentence = (const int*)d_in[0];
  const float* masks    = (const float*)d_in[1];
  const float* emb      = (const float*)d_in[2];
  const float* wih[4] = {(const float*)d_in[3],  (const float*)d_in[7],
                         (const float*)d_in[11], (const float*)d_in[15]};
  const float* whh[4] = {(const float*)d_in[4],  (const float*)d_in[8],
                         (const float*)d_in[12], (const float*)d_in[16]};
  const float* bih[4] = {(const float*)d_in[5],  (const float*)d_in[9],
                         (const float*)d_in[13], (const float*)d_in[17]};
  const float* bhh[4] = {(const float*)d_in[6],  (const float*)d_in[10],
                         (const float*)d_in[14], (const float*)d_in[18]};
  const float* wout  = (const float*)d_in[19];
  const float* bout  = (const float*)d_in[20];
  const float* trans = (const float*)d_in[21];
  const float* h0_1 = (const float*)d_in[22];
  const float* c0_1 = (const float*)d_in[23];
  const float* h0_2 = (const float*)d_in[24];
  const float* c0_2 = (const float*)d_in[25];

  char* ws = (char*)d_ws;
  size_t off = 0;
  _Float16* Xh   = (_Float16*)(ws + off); off += (size_t)LB * EE * 2;   // 16 MB
  _Float16* Hout = (_Float16*)(ws + off); off += (size_t)LB * HH * 2;   // 16 MB
  float*    Gf   = (float*)(ws + off);    off += (size_t)LB * G4 * 4;   // 64 MB
  float*    Gr   = (float*)(ws + off);    off += (size_t)LB * G4 * 4;   // 64 MB
  _Float16* WihH = (_Float16*)(ws + off); off += (size_t)4 * G4 * EE * 2;
  _Float16* WhhH = (_Float16*)(ws + off); off += (size_t)4 * G4 * HD * 2;
  float*    F    = (float*)(ws + off);    off += (size_t)LB * CCT * 4;

  convert_weights<<<3072, 256, 0, stream>>>(wih[0], wih[1], wih[2], wih[3],
                                            whh[0], whh[1], whh[2], whh[3],
                                            WihH, WhhH);
  embed_kernel<<<LB, 256, 0, stream>>>(sentence, emb, Xh);

  dim3 gg(LB / 128, G4 / 64, 2);
  size_t recShmem = (size_t)G4 * HD * 2 + 2 * (size_t)BB * HD * 2;  // 160 KB
  size_t vitShmem = (size_t)LL * 16 * CCT;                          // 96 KB

  // layer 1
  gates_gemm<<<gg, 256, 0, stream>>>(Xh, WihH, WihH + (size_t)1 * G4 * EE,
                                     bih[0], bhh[0], bih[1], bhh[1], Gf, Gr);
  lstm_rec<<<2, 256, recShmem, stream>>>(Gf, Gr, WhhH, WhhH + (size_t)1 * G4 * HD,
                                         h0_1, c0_1, Hout);
  // layer 2
  gates_gemm<<<gg, 256, 0, stream>>>(Hout, WihH + (size_t)2 * G4 * EE,
                                     WihH + (size_t)3 * G4 * EE,
                                     bih[2], bhh[2], bih[3], bhh[3], Gf, Gr);
  lstm_rec<<<2, 256, recShmem, stream>>>(Gf, Gr, WhhH + (size_t)2 * G4 * HD,
                                         WhhH + (size_t)3 * G4 * HD,
                                         h0_2, c0_2, Xh);  // out2 (f16) reuses Xh

  proj_kernel<<<LB / 8, 256, 0, stream>>>(Xh, wout, bout, F);
  viterbi_kernel<<<BB / 16, 256, vitShmem, stream>>>(F, masks, trans, (float*)d_out);
}
